// LSTMModel_103079215842
// MI455X (gfx1250) — matrix operand order repaired and checked by reading, not compiled, not run on hardware
//
#include <hip/hip_runtime.h>
#include <hip/hip_bf16.h>
#include <stdint.h>

// Problem constants (match reference: B=64, T=256, D=H=1024, L=4)
#define Bsz  64
#define Tsz  256
#define Dsz  1024
#define Hsz  1024
#define G4   4096   // 4*H gate width
#define NL   4
#define BK   32     // K-step per WMMA stage

typedef __attribute__((ext_vector_type(16))) __bf16 v16bf;
typedef __attribute__((ext_vector_type(8)))  float  v8f;

union BFrag { v16bf v; uint4 u[2]; };   // 32 bytes: one WMMA A or B fragment per lane

__device__ __forceinline__ unsigned short f2bf(float f) {
  union { float f; unsigned u; } x; x.f = f;
  unsigned r = x.u + 0x7FFFu + ((x.u >> 16) & 1u);   // round-to-nearest-even
  return (unsigned short)(r >> 16);
}

// ---- CDNA5 async-copy helpers (ASYNCcnt path) -------------------------------
__device__ __forceinline__ void async_g2l_b128(const void* gaddr, unsigned lds_byte_off) {
  // GLOBAL_LOAD_ASYNC_TO_LDS_B128 (GV mode): VDST = LDS byte address, VADDR = 64b global
  asm volatile("global_load_async_to_lds_b128 %0, %1, off"
               :: "v"(lds_byte_off), "v"(gaddr) : "memory");
}
__device__ __forceinline__ void wait_asynccnt_le2() {
#if __has_builtin(__builtin_amdgcn_s_wait_asynccnt)
  __builtin_amdgcn_s_wait_asynccnt(2);
#else
  asm volatile("s_wait_asynccnt 0x2" ::: "memory");
#endif
}
__device__ __forceinline__ void wait_asynccnt_0() {
#if __has_builtin(__builtin_amdgcn_s_wait_asynccnt)
  __builtin_amdgcn_s_wait_asynccnt(0);
#else
  asm volatile("s_wait_asynccnt 0x0" ::: "memory");
#endif
}
// Low 32 bits of a flat shared-segment pointer == wave-relative LDS byte address
__device__ __forceinline__ unsigned lds_addr32(const void* p) {
  return (unsigned)(uintptr_t)p;
}

// ---------------------------------------------------------------- conversions
__global__ void k_f32_to_bf16(const float* __restrict__ in,
                              unsigned short* __restrict__ out, int n) {
  int i = blockIdx.x * blockDim.x + threadIdx.x;
  if (i < n) out[i] = f2bf(in[i]);
}

// in: [K][N] f32 row-major  ->  out: [N][K] bf16 (column-major weights for B-fragments)
__global__ void k_transpose_bf16(const float* __restrict__ in,
                                 unsigned short* __restrict__ out, int K, int N) {
  __shared__ float tile[32][33];
  const int kb = blockIdx.x * 32, nb = blockIdx.y * 32;
  const int tx = threadIdx.x & 31, ty = threadIdx.x >> 5;   // 256 thr: ty 0..7
  #pragma unroll
  for (int r = ty; r < 32; r += 8)
    tile[r][tx] = in[(size_t)(kb + r) * N + (nb + tx)];
  __syncthreads();
  #pragma unroll
  for (int r = ty; r < 32; r += 8)
    out[(size_t)(nb + r) * K + (kb + tx)] = f2bf(tile[tx][r]);
}

__global__ void k_zero_u32(unsigned int* p) {
  if (threadIdx.x == 0 && blockIdx.x == 0) *p = 0u;
}

// ------------------------------------------------- xg = X @ WiT + bias (big GEMM)
// A: [16384][1024] bf16 row-major, WT: [4096][1024] bf16 (N-major), xg: [16384][4096] f32
// WG = 256 thr = 8 waves, WG tile 128(M) x 128(N); wave tile 32(M) x 64(N) = 2x4 frags.
// Weight tile (128N x 32K bf16 = 8 KB) is double-buffered in LDS via async copies:
// dedups the 4-way redundant B-fragment fetch across waves and overlaps copy with WMMA.
#define BROW 40   // padded row length in shorts (80 B: keeps b128 alignment)
__global__ __launch_bounds__(256, 2)
void k_gemm_xg(const unsigned short* __restrict__ A,
               const unsigned short* __restrict__ WT,
               const float* __restrict__ bias,
               float* __restrict__ xg) {
  __shared__ unsigned short ldsB[2][128][BROW];   // 20 KB
  const int lane = threadIdx.x & 31;
  const int wave = threadIdx.x >> 5;
  const int lrow = lane & 15, half = lane >> 4;
  const int Mbase   = blockIdx.x * 128 + (wave & 3) * 32;
  const int NbaseWG = blockIdx.y * 128;
  const int wn = wave >> 2;                        // N half of the WG tile
  const int Nbase = NbaseWG + wn * 64;

  // async-copy plan: 128 rows x 64 B = 512 x 16 B chunks, 2 per thread
  const int r0 = threadIdx.x >> 2;                 // rows r0, r0+64
  const int cc = (threadIdx.x & 3) * 8;            // 16 B column chunk (in shorts)
  auto issueB = [&](int buf, int k0) {
    #pragma unroll
    for (int j = 0; j < 2; ++j) {
      const int row = r0 + j * 64;
      async_g2l_b128(WT + (size_t)(NbaseWG + row) * Dsz + k0 + cc,
                     lds_addr32(&ldsB[buf][row][cc]));
    }
  };

  v8f acc[2][4];
  #pragma unroll
  for (int nf = 0; nf < 4; ++nf) {
    const float bv = bias[Nbase + nf * 16 + lrow];  // C col = lane%16
    #pragma unroll
    for (int mf = 0; mf < 2; ++mf)
      #pragma unroll
      for (int e = 0; e < 8; ++e) acc[mf][nf][e] = bv;
  }

  issueB(0, 0);
  for (int k0 = 0; k0 < Dsz; k0 += BK) {
    const int cur = (k0 / BK) & 1;
    const bool more = (k0 + BK) < Dsz;
    if (more) issueB(cur ^ 1, k0 + BK);            // prefetch next tile into other buffer

    // A fragments: direct global b128 loads (row-major, per-lane contiguous)
    BFrag a[2];
    #pragma unroll
    for (int mf = 0; mf < 2; ++mf) {
      const unsigned short* p =
          A + (size_t)(Mbase + mf * 16 + lrow) * Dsz + k0 + half * 8;
      a[mf].u[0] = *(const uint4*)p;               // K = k0+half*8 .. +7
      a[mf].u[1] = *(const uint4*)(p + 16);        // K = k0+16+half*8 .. +7
    }
    if (more)                                      // keep the A stream warm too
      __builtin_prefetch(A + (size_t)(Mbase + lrow) * Dsz + k0 + BK, 0, 1);

    if (more) wait_asynccnt_le2();                 // current tile's 2 copies done
    else      wait_asynccnt_0();
    __syncthreads();                               // ... in every wave

    BFrag b[4];
    #pragma unroll
    for (int nf = 0; nf < 4; ++nf) {               // B fragment from LDS (ds_load_b128 x2)
      const unsigned short* p = &ldsB[cur][wn * 64 + nf * 16 + lrow][half * 8];
      b[nf].u[0] = *(const uint4*)p;
      b[nf].u[1] = *(const uint4*)(p + 16);
    }
    #pragma unroll
    for (int mf = 0; mf < 2; ++mf)
      #pragma unroll
      for (int nf = 0; nf < 4; ++nf)
        acc[mf][nf] = __builtin_amdgcn_wmma_f32_16x16x32_bf16(
            false, a[mf].v, false, b[nf].v, (short)0, acc[mf][nf], false, false);
    __syncthreads();                               // done reading buf[cur] before reuse
  }

  #pragma unroll
  for (int mf = 0; mf < 2; ++mf)
    #pragma unroll
    for (int nf = 0; nf < 4; ++nf) {
      const int n = Nbase + nf * 16 + lrow;
      #pragma unroll
      for (int e = 0; e < 8; ++e) {
        const int m = Mbase + mf * 16 + half * 8 + e;   // C/D row map
        xg[(size_t)m * G4 + n] = acc[mf][nf][e];
      }
    }
}

// ------------------------------------------------- persistent recurrence scan
// 32 WGs; WG g owns h columns [g*32, g*32+32) -> computes those columns of all
// 4 gates. Per timestep: 64x128 gate GEMM tile (K=1024) via WMMA with the xg
// add fused into the accumulator init, gates -> LDS, fused LSTM cell update,
// then a device-wide phase barrier before t+1.
__global__ __launch_bounds__(256, 1)
void k_lstm_scan(const float* __restrict__ xg,            // [B*T][4096], m=b*T+t
                 const unsigned short* __restrict__ WhT,  // [4096][1024] bf16
                 unsigned short* __restrict__ Ybf,        // [B*T][1024] bf16 out (= h state)
                 float* __restrict__ C,                   // [64][1024] f32 cell state
                 float* __restrict__ Yf32,                // d_out for last layer, else null
                 unsigned int* __restrict__ bar) {
  __shared__ float gbuf[4][Bsz][32];     // 32 KB gate pre-activations
  const int lane = threadIdx.x & 31;
  const int wave = threadIdx.x >> 5;
  const int lrow = lane & 15, half = lane >> 4;
  const int hbase = blockIdx.x * 32;     // this WG's h columns
  const int mt = wave >> 1;              // batch tile 0..3 (16 rows each)
  const int nh = wave & 1;               // gate pair: 0 -> {i,f}, 1 -> {g,o}

  for (int t = 0; t < Tsz; ++t) {
    v8f acc[4];
    // init accumulators with xg tile (fuses the "+ xg_t" add)
    #pragma unroll
    for (int i = 0; i < 4; ++i) {
      const int gate = nh * 2 + (i >> 1);
      const int ncol = gate * Hsz + hbase + (i & 1) * 16 + lrow;
      #pragma unroll
      for (int e = 0; e < 8; ++e) {
        const int b = mt * 16 + half * 8 + e;
        acc[i][e] = xg[((size_t)b * Tsz + t) * G4 + ncol];
      }
    }
    if (t > 0) {   // h_{t-1} @ Wh ; h_{-1} == 0 so skip at t==0
      for (int k0 = 0; k0 < Hsz; k0 += BK) {
        BFrag a, b4[4];
        {
          const unsigned short* p =
              Ybf + ((size_t)(mt * 16 + lrow) * Tsz + (t - 1)) * Hsz + k0 + half * 8;
          a.u[0] = *(const uint4*)p;
          a.u[1] = *(const uint4*)(p + 16);
        }
        #pragma unroll
        for (int i = 0; i < 4; ++i) {
          const int gate = nh * 2 + (i >> 1);
          const int ncol = gate * Hsz + hbase + (i & 1) * 16 + lrow;
          const unsigned short* p = WhT + (size_t)ncol * Hsz + k0 + half * 8;
          b4[i].u[0] = *(const uint4*)p;
          b4[i].u[1] = *(const uint4*)(p + 16);
        }
        #pragma unroll
        for (int i = 0; i < 4; ++i)
          acc[i] = __builtin_amdgcn_wmma_f32_16x16x32_bf16(
              false, a.v, false, b4[i].v, (short)0, acc[i], false, false);
      }
    }
    // gate pre-activations -> LDS
    #pragma unroll
    for (int i = 0; i < 4; ++i) {
      const int gate = nh * 2 + (i >> 1);
      const int nloc = (i & 1) * 16 + lrow;
      #pragma unroll
      for (int e = 0; e < 8; ++e)
        gbuf[gate][mt * 16 + half * 8 + e][nloc] = acc[i][e];
    }
    __syncthreads();
    // fused LSTM cell update: 64*32 = 2048 elements, 8 per thread
    #pragma unroll
    for (int j = 0; j < 8; ++j) {
      const int idx = threadIdx.x + j * 256;
      const int b = idx >> 5, nl = idx & 31;
      const float si = 1.f / (1.f + __expf(-gbuf[0][b][nl]));
      const float sf = 1.f / (1.f + __expf(-gbuf[1][b][nl]));
      const float tg = tanhf(gbuf[2][b][nl]);
      const float so = 1.f / (1.f + __expf(-gbuf[3][b][nl]));
      const int n = hbase + nl;
      const float cp = (t == 0) ? 0.f : C[(size_t)b * Hsz + n];
      const float cn = sf * cp + si * tg;
      const float h  = so * tanhf(cn);
      C[(size_t)b * Hsz + n] = cn;
      Ybf[((size_t)b * Tsz + t) * Hsz + n] = f2bf(h);
      if (Yf32) Yf32[((size_t)b * Tsz + t) * Hsz + n] = h;
    }
    __syncthreads();
    // device-wide phase barrier: every WG must see the full h_t before t+1
    __threadfence();
    if (threadIdx.x == 0) {
      __hip_atomic_fetch_add(bar, 1u, __ATOMIC_RELEASE, __HIP_MEMORY_SCOPE_AGENT);
      const unsigned target = (unsigned)gridDim.x * (unsigned)(t + 1);
      while (__hip_atomic_load(bar, __ATOMIC_ACQUIRE, __HIP_MEMORY_SCOPE_AGENT) < target)
        __builtin_amdgcn_s_sleep(1);
    }
    __syncthreads();
  }
}

// ---------------------------------------------------------------- host driver
extern "C" void kernel_launch(void* const* d_in, const int* in_sizes, int n_in,
                              void* d_out, int out_size, void* d_ws, size_t ws_size,
                              hipStream_t stream) {
  (void)in_sizes; (void)n_in; (void)out_size; (void)ws_size;
  const float* x  = (const float*)d_in[0];   // [64,256,1024]
  const float* Wi = (const float*)d_in[1];   // [4,1024,4096]
  const float* Wh = (const float*)d_in[2];   // [4,1024,4096]
  const float* bb = (const float*)d_in[3];   // [4,4096]

  char* ws = (char*)d_ws;
  size_t off = 0;
  auto carve = [&](size_t bytes) -> void* {
    void* p = ws + off;
    off = (off + bytes + 255) & ~(size_t)255;
    return p;
  };
  // ~340 MB workspace
  float*          xg  = (float*)carve((size_t)Bsz * Tsz * G4 * 4);        // 256 MB
  unsigned short* WiT = (unsigned short*)carve((size_t)G4 * Dsz * 2);     //   8 MB
  unsigned short* WhT = (unsigned short*)carve((size_t)G4 * Hsz * 2);     //   8 MB
  unsigned short* P0  = (unsigned short*)carve((size_t)Bsz * Tsz * Hsz * 2); // 32 MB
  unsigned short* P1  = (unsigned short*)carve((size_t)Bsz * Tsz * Hsz * 2); // 32 MB
  float*          Cst = (float*)carve((size_t)Bsz * Hsz * 4);
  unsigned int*   ctr = (unsigned int*)carve(256);

  {
    const int n = Bsz * Tsz * Dsz;
    k_f32_to_bf16<<<(n + 255) / 256, 256, 0, stream>>>(x, P0, n);
  }

  unsigned short* Ain  = P0;
  unsigned short* Yout = P1;
  for (int l = 0; l < NL; ++l) {
    const dim3 tg(Dsz / 32, G4 / 32);
    k_transpose_bf16<<<tg, 256, 0, stream>>>(Wi + (size_t)l * Dsz * G4, WiT, Dsz, G4);
    k_transpose_bf16<<<tg, 256, 0, stream>>>(Wh + (size_t)l * Hsz * G4, WhT, Hsz, G4);

    const dim3 gg((Bsz * Tsz) / 128, G4 / 128);
    k_gemm_xg<<<gg, 256, 0, stream>>>(Ain, WiT, bb + (size_t)l * G4, xg);

    k_zero_u32<<<1, 64, 0, stream>>>(ctr);
    float* yf = (l == NL - 1) ? (float*)d_out : nullptr;
    k_lstm_scan<<<Hsz / 32, 256, 0, stream>>>(xg, WhT, Yout, Cst, yf, ctr);

    unsigned short* t2 = Ain; Ain = Yout; Yout = t2;   // ping-pong activations
  }
}
